// EvidentialGATLayer_83829171683613
// MI455X (gfx1250) — matrix operand order repaired and checked by reading, not compiled, not run on hardware
//
#include <hip/hip_runtime.h>

#define NNODES 50000
#define NEDGES 800000
#define INDIM 256
#define OUTDIM 128
#define HID 64
#define NCLS 3
#define ROWS_PB 64   // rows of C per block in the WMMA GEMM

typedef __attribute__((ext_vector_type(16))) __bf16        v16bf;
typedef __attribute__((ext_vector_type(8)))  float         v8f;
typedef __attribute__((ext_vector_type(8)))  unsigned int  v8u;

// ---------- helpers ----------
__device__ __forceinline__ unsigned int pack_bf2(float a, float b) {
  unsigned ua = __float_as_uint(a);
  unsigned ub = __float_as_uint(b);
  ua += 0x7FFFu + ((ua >> 16) & 1u);          // round-to-nearest-even
  ub += 0x7FFFu + ((ub >> 16) & 1u);
  return (ua >> 16) | ((ub) & 0xFFFF0000u);
}

__device__ __forceinline__ unsigned int encF(float f) {
  unsigned u = __float_as_uint(f);
  return (u & 0x80000000u) ? ~u : (u | 0x80000000u);
}
__device__ __forceinline__ float decF(unsigned e) {
  unsigned u = (e & 0x80000000u) ? (e & 0x7FFFFFFFu) : ~e;
  return __uint_as_float(u);
}
__device__ __forceinline__ void atomAddF(float* p, float v) {
  __hip_atomic_fetch_add(p, v, __ATOMIC_RELAXED, __HIP_MEMORY_SCOPE_AGENT);
}

// ---------- 0: zero x_new + init reduction scalars ----------
__global__ __launch_bounds__(256) void k_init(float* xnew, unsigned int* maxEnc, float* sumP) {
  int i = blockIdx.x * 256 + threadIdx.x;
  if (i < NNODES * OUTDIM) xnew[i] = 0.0f;
  if (i == 0) { *maxEnc = 0u; *sumP = 0.0f; }
}

// ---------- 1: W [K=256][N=128] f32  ->  Bt bf16 packed, N-major [128][128 pairs] ----------
__global__ __launch_bounds__(256) void k_cvtW(const float* __restrict__ W,
                                              unsigned int* __restrict__ Bt) {
  int t = blockIdx.x * 256 + threadIdx.x;       // 16384 packed elements
  if (t >= OUTDIM * (INDIM / 2)) return;
  int n  = t >> 7;                               // 0..127
  int kp = t & 127;                              // K pair index
  float a = W[(2 * kp)     * OUTDIM + n];
  float b = W[(2 * kp + 1) * OUTDIM + n];
  Bt[t] = pack_bf2(a, b);
}

// ---------- 2: Wh = x @ W via bf16 WMMA ----------
// block = 256 threads (8 waves). Block computes ROWS_PB rows x all 128 cols.
// Wave w owns the 16-col strip at 16*w. B fragments (8 k-steps) are preloaded
// into registers ONCE per wave, then reused across ROWS_PB/16 row tiles.
__global__ __launch_bounds__(256) void k_gemm(const float* __restrict__ x,
                                              const unsigned int* __restrict__ Bt,
                                              float* __restrict__ Wh) {
  __shared__ __align__(16) unsigned int ldsA[ROWS_PB * 128];   // 64 rows x 128 pairs = 32KB
  const int rowBase = blockIdx.x * ROWS_PB;
  const int tid = threadIdx.x;

  // Stage A tile: ROWS_PB x 256 f32 -> packed bf16 pairs in LDS (coalesced float2 loads)
  const float2* xsrc = (const float2*)(x + (size_t)rowBase * INDIM);
  const int pairLimit = (NNODES - rowBase) * (INDIM / 2);      // valid pairs in this block
  #pragma unroll
  for (int p = tid; p < ROWS_PB * 128; p += 256) {
    if (p < pairLimit) {
      float2 v = xsrc[p];
      ldsA[p] = pack_bf2(v.x, v.y);
    }
  }
  __syncthreads();

  const int wave = tid >> 5;
  const int lane = tid & 31;
  const int half = lane >> 4;        // 0: K-low half, 1: K-high half
  const int m    = lane & 15;
  const int n    = wave * 16 + m;    // output column

  // Preload B fragments (32x16 bf16 each) for all 8 K-steps: 8 contiguous pairs/lane.
  v8u bu[8];
  #pragma unroll
  for (int kk = 0; kk < 8; ++kk) {
    int pb = n * 128 + kk * 16 + half * 8;
    uint4 b0 = *(const uint4*)&Bt[pb];
    uint4 b1 = *(const uint4*)&Bt[pb + 4];
    bu[kk][0] = b0.x; bu[kk][1] = b0.y; bu[kk][2] = b0.z; bu[kk][3] = b0.w;
    bu[kk][4] = b1.x; bu[kk][5] = b1.y; bu[kk][6] = b1.z; bu[kk][7] = b1.w;
  }

  #pragma unroll
  for (int rt = 0; rt < ROWS_PB / 16; ++rt) {
    if (rowBase + rt * 16 >= NNODES) break;     // 50000 is 16-aligned: all-or-nothing tiles
    v8f c = {};
    #pragma unroll
    for (int kk = 0; kk < 8; ++kk) {
      // A frag (16x32 bf16): lanes 0-15 take K {0..15}, lanes 16-31 {16..31}
      int pa = (rt * 16 + m) * 128 + kk * 16 + half * 8;
      uint4 a0 = *(const uint4*)&ldsA[pa];       // 4 pairs: K base+0..7
      uint4 a1 = *(const uint4*)&ldsA[pa + 4];   // 4 pairs: K base+8..15
      v8u au;
      au[0] = a0.x; au[1] = a0.y; au[2] = a0.z; au[3] = a0.w;
      au[4] = a1.x; au[5] = a1.y; au[6] = a1.z; au[7] = a1.w;
      c = __builtin_amdgcn_wmma_f32_16x16x32_bf16(
          false, __builtin_bit_cast(v16bf, au),
          false, __builtin_bit_cast(v16bf, bu[kk]),
          (short)0, c, false, false);
    }
    // C layout: VGPR v -> M = v + half*8, lane%16 -> N
    float* wrow = Wh + (size_t)(rowBase + rt * 16 + half * 8) * OUTDIM + n;
    #pragma unroll
    for (int v = 0; v < 8; ++v) wrow[(size_t)v * OUTDIM] = c[v];
  }
}

// ---------- 3: s_src = Wh@a[:128], s_tgt = Wh@a[128:], one wave per node ----------
__global__ __launch_bounds__(256) void k_svec(const float* __restrict__ Wh,
                                              const float* __restrict__ a,
                                              float* __restrict__ s1,
                                              float* __restrict__ s2) {
  int node = blockIdx.x * 8 + (threadIdx.x >> 5);
  if (node >= NNODES) return;
  int lane = threadIdx.x & 31;
  float4 w  = *(const float4*)&Wh[(size_t)node * OUTDIM + lane * 4];
  float4 a1 = *(const float4*)&a[lane * 4];
  float4 a2 = *(const float4*)&a[OUTDIM + lane * 4];
  float d1 = w.x * a1.x + w.y * a1.y + w.z * a1.z + w.w * a1.w;
  float d2 = w.x * a2.x + w.y * a2.y + w.z * a2.z + w.w * a2.w;
  #pragma unroll
  for (int off = 16; off; off >>= 1) {
    d1 += __shfl_down(d1, off, 32);
    d2 += __shfl_down(d2, off, 32);
  }
  if (lane == 0) { s1[node] = d1; s2[node] = d2; }
}

// ---------- 4: edge scores = leaky_relu(s_src[src]+s_tgt[tgt]); global max ----------
__global__ __launch_bounds__(256) void k_score(const int* __restrict__ ei,
                                               const float* __restrict__ s1,
                                               const float* __restrict__ s2,
                                               float* __restrict__ scores,
                                               unsigned int* __restrict__ maxEnc) {
  int e = blockIdx.x * 256 + threadIdx.x;
  float sc = -3.4e38f;
  if (e < NEDGES) {
    int s = ei[e];
    int t = ei[NEDGES + e];
    float z = s1[s] + s2[t];
    sc = z > 0.0f ? z : 0.2f * z;
    scores[e] = sc;
  }
  #pragma unroll
  for (int off = 16; off; off >>= 1) sc = fmaxf(sc, __shfl_down(sc, off, 32));
  if ((threadIdx.x & 31) == 0) atomicMax(maxEnc, encF(sc));
}

// ---------- 5: w = exp(score - max) in place; global sum ----------
__global__ __launch_bounds__(256) void k_exps(float* __restrict__ scores,
                                              const unsigned int* __restrict__ maxEnc,
                                              float* __restrict__ sumP) {
  int e = blockIdx.x * 256 + threadIdx.x;
  float mv = decF(*maxEnc);
  float w = 0.0f;
  if (e < NEDGES) {
    w = __expf(scores[e] - mv);
    scores[e] = w;
  }
  #pragma unroll
  for (int off = 16; off; off >>= 1) w += __shfl_down(w, off, 32);
  if ((threadIdx.x & 31) == 0) atomAddF(sumP, w);
}

// ---------- 6: scatter x_new[tgt] += alpha * Wh[src]; one wave per edge ----------
__global__ __launch_bounds__(256) void k_scatter(const int* __restrict__ ei,
                                                 const float* __restrict__ w,
                                                 const float* __restrict__ sumP,
                                                 const float* __restrict__ Wh,
                                                 float* __restrict__ xnew) {
  int e = blockIdx.x * 8 + (threadIdx.x >> 5);
  if (e >= NEDGES) return;
  int lane  = threadIdx.x & 31;
  float alpha = w[e] / sumP[0];
  int s = ei[e];
  int t = ei[NEDGES + e];
  float4 v = *(const float4*)&Wh[(size_t)s * OUTDIM + lane * 4];
  float* dst = &xnew[(size_t)t * OUTDIM + lane * 4];
  atomAddF(dst + 0, alpha * v.x);
  atomAddF(dst + 1, alpha * v.y);
  atomAddF(dst + 2, alpha * v.z);
  atomAddF(dst + 3, alpha * v.w);
}

// ---------- 7: MLP head: evidence = softplus(relu(x_new@W1+b1)@W2+b2)+1 ----------
__global__ __launch_bounds__(64) void k_head(const float* __restrict__ xnew,
                                             const float* __restrict__ W1,
                                             const float* __restrict__ b1,
                                             const float* __restrict__ W2,
                                             const float* __restrict__ b2,
                                             float* __restrict__ ev) {
  __shared__ float h[HID];
  int node = blockIdx.x;
  int j = threadIdx.x;
  const float* row = xnew + (size_t)node * OUTDIM;
  float acc = b1[j];
  #pragma unroll 8
  for (int k = 0; k < OUTDIM; ++k) acc += row[k] * W1[k * HID + j];
  h[j] = acc > 0.0f ? acc : 0.0f;
  __syncthreads();
  if (j < NCLS) {
    float a2 = b2[j];
    #pragma unroll 8
    for (int k = 0; k < HID; ++k) a2 += h[k] * W2[k * NCLS + j];
    float sp = (a2 > 20.0f) ? a2 : log1pf(__expf(a2));
    ev[(size_t)node * NCLS + j] = sp + 1.0f;
  }
}

// ---------- launch ----------
extern "C" void kernel_launch(void* const* d_in, const int* in_sizes, int n_in,
                              void* d_out, int out_size, void* d_ws, size_t ws_size,
                              hipStream_t stream) {
  const float* x  = (const float*)d_in[0];
  const int*   ei = (const int*)d_in[1];   // edge_index [2, E]
  const float* W  = (const float*)d_in[2];
  const float* a  = (const float*)d_in[3];
  const float* W1 = (const float*)d_in[4];
  const float* b1 = (const float*)d_in[5];
  const float* W2 = (const float*)d_in[6];
  const float* b2 = (const float*)d_in[7];
  (void)in_sizes; (void)n_in; (void)out_size; (void)ws_size;

  float* out  = (float*)d_out;
  float* xnew = out;                                   // [NNODES * OUTDIM]
  float* ev   = out + (size_t)NNODES * OUTDIM;         // [NNODES * NCLS]

  // workspace layout (4-byte units)
  unsigned int* ws32 = (unsigned int*)d_ws;
  float*        Wh     = (float*)ws32;                             // 6,400,000
  unsigned int* Bt     = ws32 + (size_t)NNODES * OUTDIM;           // 16,384
  float*        s1     = (float*)(Bt + OUTDIM * (INDIM / 2));      // 50,000
  float*        s2     = s1 + NNODES;                              // 50,000
  float*        scores = s2 + NNODES;                              // 800,000
  unsigned int* maxEnc = (unsigned int*)(scores + NEDGES);         // 1
  float*        sumP   = (float*)(maxEnc + 1);                     // 1

  k_init   <<<(NNODES * OUTDIM + 255) / 256, 256, 0, stream>>>(xnew, maxEnc, sumP);
  k_cvtW   <<<(OUTDIM * (INDIM / 2) + 255) / 256, 256, 0, stream>>>(W, Bt);
  k_gemm   <<<(NNODES + ROWS_PB - 1) / ROWS_PB, 256, 0, stream>>>(x, Bt, Wh);
  k_svec   <<<(NNODES + 7) / 8, 256, 0, stream>>>(Wh, a, s1, s2);
  k_score  <<<(NEDGES + 255) / 256, 256, 0, stream>>>(ei, s1, s2, scores, maxEnc);
  k_exps   <<<(NEDGES + 255) / 256, 256, 0, stream>>>(scores, maxEnc, sumP);
  k_scatter<<<(NEDGES + 7) / 8, 256, 0, stream>>>(ei, scores, sumP, Wh, xnew);
  k_head   <<<NNODES, 64, 0, stream>>>(xnew, W1, b1, W2, b2, ev);
}
